// UNetCustom_18588618457467
// MI455X (gfx1250) — matrix operand order repaired and checked by reading, not compiled, not run on hardware
//
#include <hip/hip_runtime.h>

typedef _Float16 half_t;
typedef __attribute__((ext_vector_type(16))) _Float16 v16h;
typedef __attribute__((ext_vector_type(8)))  float    v8f;
typedef __attribute__((ext_vector_type(4)))  float    v4f;
typedef __attribute__((ext_vector_type(4)))  unsigned v4u;
typedef __attribute__((ext_vector_type(8)))  unsigned v8u;

#define NB   16
#define C1   18
#define C2   36
#define HW   65536      // 256*256
#define CNT  (16.0f*65536.0f)

// ---------------------------------------------------------------- conv1 (direct, fp32; K=48 too small for WMMA, HBM-bound)
__global__ void __launch_bounds__(256) k_conv1(const float* __restrict__ x,
                                               const float* __restrict__ w,
                                               float* __restrict__ y1) {
  __shared__ float ws[C1 * 48];
  for (int i = threadIdx.x; i < C1 * 48; i += 256) ws[i] = w[i];
  __syncthreads();
  int p  = blockIdx.x * 256 + threadIdx.x;      // 0 .. 1048575 (n*65536 + oh*256 + ow)
  int n  = p >> 16;
  int hw = p & 65535;
  int oh = hw >> 8, ow = hw & 255;
  int ih0 = oh * 2 - 2, iw0 = ow * 2 - 2;
  float xv[48];
#pragma unroll
  for (int c = 0; c < 3; ++c)
#pragma unroll
    for (int ky = 0; ky < 4; ++ky)
#pragma unroll
      for (int kx = 0; kx < 4; ++kx) {
        int ih = ih0 + ky, iw = iw0 + kx;
        float v = 0.f;
        if (ih >= 0 && ih < 510 && iw >= 0 && iw < 510)
          v = x[((size_t)(n * 3 + c) * 510 + ih) * 510 + iw];
        xv[c * 16 + ky * 4 + kx] = v;
      }
  for (int o = 0; o < C1; ++o) {
    float a = 0.f;
#pragma unroll
    for (int i = 0; i < 48; ++i) a = fmaf(ws[o * 48 + i], xv[i], a);
    y1[((size_t)(n * C1 + o) << 16) + hw] = a;   // bias dropped: cancels in train-mode BN
  }
}

// ---------------------------------------------------------------- per-channel sum / sumsq partials (deterministic 2-stage)
__global__ void __launch_bounds__(256) k_bnstats(const float* __restrict__ y,
                                                 float* __restrict__ psum,
                                                 float* __restrict__ psq, int C) {
  int c = blockIdx.y;
  __shared__ float r0[256], r1[256];
  float a = 0.f, b = 0.f;
  int base = blockIdx.x * 16384;
  for (int t = threadIdx.x; t < 16384; t += 256) {
    int q = base + t;
    int n = q >> 16, p = q & 65535;
    float v = y[((size_t)(n * C + c) << 16) + p];
    a += v; b += v * v;
  }
  r0[threadIdx.x] = a; r1[threadIdx.x] = b; __syncthreads();
  for (int s = 128; s > 0; s >>= 1) {
    if (threadIdx.x < s) { r0[threadIdx.x] += r0[threadIdx.x + s]; r1[threadIdx.x] += r1[threadIdx.x + s]; }
    __syncthreads();
  }
  if (threadIdx.x == 0) { psum[c * 64 + blockIdx.x] = r0[0]; psq[c * 64 + blockIdx.x] = r1[0]; }
}

__global__ void k_reduce64(const float* __restrict__ src, float* __restrict__ dst) {
  __shared__ float red[64];
  red[threadIdx.x] = src[blockIdx.x * 64 + threadIdx.x];
  __syncthreads();
  for (int s = 32; s > 0; s >>= 1) { if (threadIdx.x < s) red[threadIdx.x] += red[threadIdx.x + s]; __syncthreads(); }
  if (threadIdx.x == 0) dst[blockIdx.x] = red[0];
}

// ---------------------------------------------------------------- BN + leaky (in/out of place) + per-(n,c) SE partial sums
__global__ void __launch_bounds__(256) k_bnapply(const float* __restrict__ src, float* __restrict__ dst,
                                                 const float* __restrict__ bsum, const float* __restrict__ bsq,
                                                 const float* __restrict__ g, const float* __restrict__ bb,
                                                 float* __restrict__ separt, int C) {
  int nc = blockIdx.y;
  int c  = nc % C;
  float mean = bsum[c] * (1.f / CNT);
  float var  = bsq[c] * (1.f / CNT) - mean * mean;
  float scl  = g[c] * rsqrtf(var + 1e-5f);
  float sh   = bb[c] - mean * scl;
  size_t base = ((size_t)nc << 16) + (size_t)blockIdx.x * 1024;
  float acc = 0.f;
  for (int t = threadIdx.x; t < 1024; t += 256) {
    float v = fmaf(src[base + t], scl, sh);
    v = v > 0.f ? v : 0.1f * v;
    dst[base + t] = v;
    acc += v;
  }
  __shared__ float red[256];
  red[threadIdx.x] = acc; __syncthreads();
  for (int s = 128; s > 0; s >>= 1) { if (threadIdx.x < s) red[threadIdx.x] += red[threadIdx.x + s]; __syncthreads(); }
  if (threadIdx.x == 0) separt[nc * 64 + blockIdx.x] = red[0];
}

// ---------------------------------------------------------------- SE gate: s = sigmoid(relu(mean@w1ᵀ)@w2ᵀ), one tiny block
__global__ void k_sescale(const float* __restrict__ sesum, const float* __restrict__ w1,
                          const float* __restrict__ w2, float* __restrict__ sout, int C, int Hd) {
  __shared__ float sm[NB * C2];
  __shared__ float hh[NB * 2];
  int tid = threadIdx.x;                 // blockDim == 16*C
  int n = tid / C, c = tid % C;
  sm[n * C + c] = sesum[n * C + c] * (1.f / 65536.f);
  __syncthreads();
  if (tid < NB * Hd) {
    int nn = tid / Hd, j = tid % Hd;
    float a = 0.f;
    for (int cc = 0; cc < C; ++cc) a += sm[nn * C + cc] * w1[j * C + cc];
    hh[nn * 2 + j] = a > 0.f ? a : 0.f;
  }
  __syncthreads();
  float z = 0.f;
  for (int j = 0; j < Hd; ++j) z += hh[n * 2 + j] * w2[c * Hd + j];
  sout[n * C + c] = 1.f / (1.f + __expf(-z));
}

// ---------------------------------------------------------------- SE1-scaled x1 -> fp16 CHW (pure elementwise, coalesced)
__global__ void __launch_bounds__(256) k_packA(const float* __restrict__ x1, const float* __restrict__ s1,
                                               half_t* __restrict__ x1h) {
  int i = blockIdx.x * 256 + threadIdx.x;     // CHW, same layout as x1
  if (i >= NB * C1 * HW) return;
  x1h[i] = (half_t)(x1[i] * s1[i >> 16]);
}

// ---------------------------------------------------------------- pack conv2 weights into exact WMMA B-fragment layout
// B (32x16 f16): lane<16 -> N=lane, half h -> K=(h&7)+((h>>3)<<4) ; lane>=16 -> N=lane-16, K=that+8   (ISA 7.12.2)
__global__ void k_packB(const float* __restrict__ w2c, half_t* __restrict__ bfr) {
  int i = blockIdx.x * 256 + threadIdx.x;     // 6 ksteps * 3 ntiles * 32 lanes * 16 halves = 9216
  if (i >= 9216) return;
  int h = i & 15, lane = (i >> 4) & 31, nt = (i >> 9) % 3, kt = i / 1536;
  int N = nt * 16 + (lane & 15);
  int K = kt * 32 + (h & 7) + ((h >> 3) << 4) + ((lane < 16) ? 0 : 8);
  half_t v = (half_t)0.f;
  if (N < C2 && K < 162) {                     // K = (ky*3+kx)*18 + c
    int dydx = K / 18, c = K - dydx * 18, ky = dydx / 3, kx = dydx - ky * 3;
    v = (half_t)w2c[((N * C1 + c) * 3 + ky) * 3 + kx];
  }
  bfr[i] = v;
}

// ---------------------------------------------------------------- A-fragment builder: K fully compile-time ->
// ds_load_u16 with immediate offsets off one per-lane base. KB = 0 (lanes 0-15) or 8 (lanes 16-31).
template<int KT, int KB>
__device__ __forceinline__ v16h mk_a(const half_t* __restrict__ Ab) {
  v16h a;
#pragma unroll
  for (int h = 0; h < 16; ++h) {
    const int K = KT * 32 + (h & 7) + ((h >> 3) << 4) + KB;   // ISA A-16x32-f16 layout
    half_t v = (half_t)0.f;
    if (K < 162) {
      const int dydx = K / 18, c = K - dydx * 18, ky = dydx / 3, kx = dydx - ky * 3;
      v = Ab[(ky * 18 + c) * 80 + kx + 7];                    // constant offset, base carries the pixel
    }
    a[h] = v;
  }
  return a;
}

// ---------------------------------------------------------------- conv2 as implicit GEMM with v_wmma_f32_16x16x32_f16
// block = 128 (4 waves); wave owns 16 pixels x 48 channels (3 N-tiles); K = 192 (6 steps), zero-padded past 162.
// B fragments arrive via the Tensor Data Mover; activation halo rows via global_load_async_to_lds_b128.
__global__ void __launch_bounds__(128) k_conv2(const half_t* __restrict__ x1h,
                                               const half_t* __restrict__ bfr,
                                               float* __restrict__ y2) {
  __shared__ __align__(16) half_t Bt[9216];          // 18 KB of weight fragments
  __shared__ __align__(16) half_t At[3 * 18 * 80];   // 54 halo rows x 80 halves (iw = ow0-8 .. ow0+71)
  int tid = threadIdx.x, lane = tid & 31, wave = tid >> 5;
  int n = blockIdx.z, oh = blockIdx.y, bx = blockIdx.x, ow0 = bx * 64;

  // --- TDM: one descriptor pulls the whole 18 KB fragment tile into LDS (TENSORcnt path)
  if (wave == 0) {
    unsigned long long ga = (unsigned long long)bfr;
    v4u g0;
    g0[0] = 1u;                                            // count=1
    g0[1] = (unsigned)(unsigned long long)&Bt[0];          // lds_addr
    g0[2] = (unsigned)ga;                                  // global_addr[31:0]
    g0[3] = (unsigned)((ga >> 32) & 0x01ffffffu) | (2u << 30);  // global_addr[56:32] | type=2
    v8u g1;
    g1[0] = 3u << 16;          // data_size = 8 bytes
    g1[1] = 2304u << 16;       // tensor_dim0 = 2304 (x 8B = 18432 B)
    g1[2] = 1u << 16;          // tensor_dim1 = 1
    g1[3] = 2304u << 16;       // tile_dim0 = 2304
    g1[4] = 1u;                // tile_dim1 = 1, tile_dim2 = 0
    g1[5] = 2304u;             // tensor_dim0_stride
    g1[6] = 0u;
    g1[7] = 0u;
    asm volatile("tensor_load_to_lds %0, %1" :: "s"(g0), "s"(g1) : "memory");
  }

  // --- activation halo rows: 10 x b128 async bursts per (ky,c) row, 16B-aligned start at ow0-8
  {
    unsigned atb = (unsigned)(unsigned long long)&At[0];
    for (int row = tid; row < 54; row += 128) {
      int ky = row / 18, c = row - ky * 18;
      int ih = oh + ky - 1;
      unsigned lrow = atb + row * 160;
      if (ih >= 0 && ih < 256) {
        unsigned long long g =
            (unsigned long long)(x1h + (((size_t)(n * C1 + c) << 16) + ih * 256 + ow0 - 8));
#pragma unroll
        for (int j = 0; j < 10; ++j)
          asm volatile("global_load_async_to_lds_b128 %0, %1, off"
                       :: "v"(lrow + j * 16u), "v"(g + j * 16u) : "memory");
      } else {
        v4u* zp = (v4u*)(At + row * 80);
#pragma unroll
        for (int j = 0; j < 10; ++j) zp[j] = (v4u){0u, 0u, 0u, 0u};
      }
    }
  }
  asm volatile("s_wait_asynccnt 0" ::: "memory");
  __builtin_amdgcn_s_wait_tensorcnt(0);
  __syncthreads();
  // --- border fixups (iw = -1 lives at wl 7; iw = 256 at wl 72)
  if (bx == 0) for (int row = tid; row < 54; row += 128) At[row * 80 + 7]  = (half_t)0.f;
  if (bx == 3) for (int row = tid; row < 54; row += 128) At[row * 80 + 72] = (half_t)0.f;
  __syncthreads();

  // --- build all 6 A fragments with compile-time LDS offsets (half-wave uniform branch, reconverges pre-WMMA)
  const half_t* Ab = At + (wave * 16 + (lane & 15));   // per-lane pixel base
  v16h Afr[6];
  if (lane < 16) {
    Afr[0] = mk_a<0, 0>(Ab); Afr[1] = mk_a<1, 0>(Ab); Afr[2] = mk_a<2, 0>(Ab);
    Afr[3] = mk_a<3, 0>(Ab); Afr[4] = mk_a<4, 0>(Ab); Afr[5] = mk_a<5, 0>(Ab);
  } else {
    Afr[0] = mk_a<0, 8>(Ab); Afr[1] = mk_a<1, 8>(Ab); Afr[2] = mk_a<2, 8>(Ab);
    Afr[3] = mk_a<3, 8>(Ab); Afr[4] = mk_a<4, 8>(Ab); Afr[5] = mk_a<5, 8>(Ab);
  }

  v8f a0 = {}, a1 = {}, a2 = {};
  const v16h* Bv = (const v16h*)Bt;
#pragma unroll
  for (int kt = 0; kt < 6; ++kt) {
    v16h b0 = Bv[(kt * 3 + 0) * 32 + lane];
    v16h b1 = Bv[(kt * 3 + 1) * 32 + lane];
    v16h b2 = Bv[(kt * 3 + 2) * 32 + lane];
    a0 = __builtin_amdgcn_wmma_f32_16x16x32_f16(false, Afr[kt], false, b0, (short)0, a0, false, false);
    a1 = __builtin_amdgcn_wmma_f32_16x16x32_f16(false, Afr[kt], false, b1, (short)0, a1, false, false);
    a2 = __builtin_amdgcn_wmma_f32_16x16x32_f16(false, Afr[kt], false, b2, (short)0, a2, false, false);
  }

  // --- D layout: lane -> column N=lane&15 (lanes 16-31 rows +8), vgpr r -> row M=r.
  //     8 rows = 8 consecutive pixels in one channel row -> two 16B stores per N-tile.
  int mbase = (lane < 16) ? 0 : 8;
  int chl = lane & 15;
  size_t po = (size_t)oh * 256 + (ow0 + wave * 16 + mbase);
  {
    size_t i0 = ((size_t)(n * C2 + chl) << 16) + po;
    *(v4f*)(y2 + i0)     = (v4f){a0[0], a0[1], a0[2], a0[3]};
    *(v4f*)(y2 + i0 + 4) = (v4f){a0[4], a0[5], a0[6], a0[7]};
    size_t i1 = ((size_t)(n * C2 + 16 + chl) << 16) + po;
    *(v4f*)(y2 + i1)     = (v4f){a1[0], a1[1], a1[2], a1[3]};
    *(v4f*)(y2 + i1 + 4) = (v4f){a1[4], a1[5], a1[6], a1[7]};
    if (chl < 4) {
      size_t i2 = ((size_t)(n * C2 + 32 + chl) << 16) + po;
      *(v4f*)(y2 + i2)     = (v4f){a2[0], a2[1], a2[2], a2[3]};
      *(v4f*)(y2 + i2 + 4) = (v4f){a2[4], a2[5], a2[6], a2[7]};
    }
  }
}

// ---------------------------------------------------------------- apply SE2 gate to x2 in place (d_out)
__global__ void __launch_bounds__(256) k_scale_x2(float* __restrict__ x2, const float* __restrict__ s2) {
  size_t i = (size_t)blockIdx.x * 256 + threadIdx.x;
  if (i >= (size_t)NB * C2 * HW) return;
  x2[i] *= s2[(int)(i >> 16)];
}

// ---------------------------------------------------------------- review branch: per-(n,o) spatial sums of relu(x1@Wᵀ+b)
// (DWT/IDWT are linear & inverse -> recon = x2 + reviewed; only the pooled sums are needed)
__global__ void __launch_bounds__(256) k_review(const float* __restrict__ x1, const float* __restrict__ rw,
                                                const float* __restrict__ rb, float* __restrict__ rpart) {
  __shared__ float w[C2 * C1];
  __shared__ float bsh[C2];
  __shared__ float red[256];
  for (int i = threadIdx.x; i < C2 * C1; i += 256) w[i] = rw[i];
  if (threadIdx.x < C2) bsh[threadIdx.x] = rb[threadIdx.x];
  __syncthreads();
  int n = blockIdx.y;
  int p0 = blockIdx.x * 1024;
  float r[C2];
#pragma unroll
  for (int o = 0; o < C2; ++o) r[o] = 0.f;
  for (int t = threadIdx.x; t < 1024; t += 256) {
    int p = p0 + t;
    float xv[C1];
#pragma unroll
    for (int c = 0; c < C1; ++c) xv[c] = x1[((size_t)(n * C1 + c) << 16) + p];
#pragma unroll
    for (int o = 0; o < C2; ++o) {
      float d = bsh[o];
#pragma unroll
      for (int c = 0; c < C1; ++c) d = fmaf(w[o * C1 + c], xv[c], d);
      r[o] += d > 0.f ? d : 0.f;
    }
  }
  for (int o = 0; o < C2; ++o) {
    red[threadIdx.x] = r[o]; __syncthreads();
    for (int s = 128; s > 0; s >>= 1) { if (threadIdx.x < s) red[threadIdx.x] += red[threadIdx.x + s]; __syncthreads(); }
    if (threadIdx.x == 0) rpart[(n * C2 + o) * 64 + blockIdx.x] = red[0];
    __syncthreads();
  }
}

// ---------------------------------------------------------------- head: out = ((Σx2_pre·s2 + Σrev)/HW) @ final_wᵀ + b
__global__ void k_final(const float* __restrict__ se2sum, const float* __restrict__ s2,
                        const float* __restrict__ rsum, const float* __restrict__ fw,
                        const float* __restrict__ fb, float* __restrict__ out) {
  int t = threadIdx.x;
  if (t >= 32) return;
  int n = t >> 1, j = t & 1;
  float a = 0.f;
  for (int c = 0; c < C2; ++c) {
    float pooled = (se2sum[n * C2 + c] * s2[n * C2 + c] + rsum[n * C2 + c]) * (1.f / 65536.f);
    a = fmaf(pooled, fw[j * C2 + c], a);
  }
  out[n * 2 + j] = a + fb[j];
}

// ================================================================ host
extern "C" void kernel_launch(void* const* d_in, const int* in_sizes, int n_in,
                              void* d_out, int out_size, void* d_ws, size_t ws_size,
                              hipStream_t stream) {
  (void)in_sizes; (void)n_in; (void)out_size; (void)ws_size;
  const float* x        = (const float*)d_in[0];
  const float* conv1_w  = (const float*)d_in[1];
  const float* bn1_g    = (const float*)d_in[3];
  const float* bn1_b    = (const float*)d_in[4];
  const float* se1_w1   = (const float*)d_in[5];
  const float* se1_w2   = (const float*)d_in[6];
  const float* conv2_w  = (const float*)d_in[7];
  const float* bn2_g    = (const float*)d_in[9];
  const float* bn2_b    = (const float*)d_in[10];
  const float* se2_w1   = (const float*)d_in[11];
  const float* se2_w2   = (const float*)d_in[12];
  const float* review_w = (const float*)d_in[13];
  const float* review_b = (const float*)d_in[14];
  const float* final_w  = (const float*)d_in[15];
  const float* final_b  = (const float*)d_in[16];

  const size_t E1 = (size_t)NB * C1 * HW;   // 18,874,368
  char* wsb = (char*)d_ws;
  float*  y1  = (float*)wsb;  wsb += E1 * 4;
  float*  x1  = (float*)wsb;  wsb += E1 * 4;
  half_t* x1h = (half_t*)wsb; wsb += E1 * 2 + 64;   // +64B slack for aligned halo over-read at buffer end
  half_t* bfr = (half_t*)wsb; wsb += 9216 * 2;
  float* a = (float*)wsb;
  float* bn1ps = a; a += C1 * 64;
  float* bn1pq = a; a += C1 * 64;
  float* bn1sum = a; a += C1;
  float* bn1sq  = a; a += C1;
  float* se1p   = a; a += NB * C1 * 64;
  float* se1sum = a; a += NB * C1;
  float* s1     = a; a += NB * C1;
  float* bn2ps  = a; a += C2 * 64;
  float* bn2pq  = a; a += C2 * 64;
  float* bn2sum = a; a += C2;
  float* bn2sq  = a; a += C2;
  float* se2p   = a; a += NB * C2 * 64;
  float* se2sum = a; a += NB * C2;
  float* s2     = a; a += NB * C2;
  float* rp     = a; a += NB * C2 * 64;
  float* rsum   = a; a += NB * C2;

  float* x2out = (float*)d_out;                      // 16*36*256*256
  float* outv  = x2out + (size_t)NB * C2 * HW;       // 16*2

  // stage 1: conv1 -> BN1 stats -> BN1+leaky (x1) -> SE1 gate -> fp16 CHW input for conv2
  k_conv1<<<4096, 256, 0, stream>>>(x, conv1_w, y1);
  k_bnstats<<<dim3(64, C1), 256, 0, stream>>>(y1, bn1ps, bn1pq, C1);
  k_reduce64<<<C1, 64, 0, stream>>>(bn1ps, bn1sum);
  k_reduce64<<<C1, 64, 0, stream>>>(bn1pq, bn1sq);
  k_bnapply<<<dim3(64, NB * C1), 256, 0, stream>>>(y1, x1, bn1sum, bn1sq, bn1_g, bn1_b, se1p, C1);
  k_reduce64<<<NB * C1, 64, 0, stream>>>(se1p, se1sum);
  k_sescale<<<1, NB * C1, 0, stream>>>(se1sum, se1_w1, se1_w2, s1, C1, 1);
  k_packA<<<(int)((E1 + 255) / 256), 256, 0, stream>>>(x1, s1, x1h);

  // stage 2: conv2 via WMMA (raw y2 straight into d_out) -> BN2 -> SE2 gate in place
  k_packB<<<36, 256, 0, stream>>>(conv2_w, bfr);
  k_conv2<<<dim3(4, 256, NB), 128, 0, stream>>>(x1h, bfr, x2out);
  k_bnstats<<<dim3(64, C2), 256, 0, stream>>>(x2out, bn2ps, bn2pq, C2);
  k_reduce64<<<C2, 64, 0, stream>>>(bn2ps, bn2sum);
  k_reduce64<<<C2, 64, 0, stream>>>(bn2pq, bn2sq);
  k_bnapply<<<dim3(64, NB * C2), 256, 0, stream>>>(x2out, x2out, bn2sum, bn2sq, bn2_g, bn2_b, se2p, C2);
  k_reduce64<<<NB * C2, 64, 0, stream>>>(se2p, se2sum);
  k_sescale<<<1, NB * C2, 0, stream>>>(se2sum, se2_w1, se2_w2, s2, C2, 2);
  k_scale_x2<<<(int)(((size_t)NB * C2 * HW + 255) / 256), 256, 0, stream>>>(x2out, s2);

  // stage 3: review pooled sums + head (wavelet stage collapsed by linearity)
  k_review<<<dim3(64, NB), 256, 0, stream>>>(x1, review_w, review_b, rp);
  k_reduce64<<<NB * C2, 64, 0, stream>>>(rp, rsum);
  k_final<<<1, 32, 0, stream>>>(se2sum, s2, rsum, final_w, final_b, outv);
}
